// MultiscaleAttention_43267500539980
// MI455X (gfx1250) — matrix operand order repaired and checked by reading, not compiled, not run on hardware
//
#include <hip/hip_runtime.h>
#include <hip/hip_bf16.h>

typedef __attribute__((ext_vector_type(16))) _Float16 v16h;
typedef __attribute__((ext_vector_type(8)))  float    v8f;

#define B_SZ   4
#define T_IN   8
#define H_IN   56
#define W_IN   56
#define N_IN   25089           // 1 + 8*56*56
#define ODIM   192
#define HDIM   96
#define QH     28
#define QW     28
#define QN     6273            // 1 + 8*28*28
#define KH     7
#define KW     7
#define KN     393             // 1 + 8*7*7
#define KNP    416             // 13*32
#define MTOT1  (B_SZ*N_IN)     // 100356
#define MT1    6273            // ceil(100356/16)
#define MTOT2  (B_SZ*QN)       // 25092
#define MT2    1569            // ceil(25092/16)
#define QTILES 393             // ceil(6273/16)
#define NTQ    26              // KNP/16
#define KSV    13              // KNP/32
#define SCALE_F 0.10206207261596575f  // 1/sqrt(96)

// fragment sizes (f16 elements)
#define XSW_ELEMS   ((size_t)MT1 * 3 * 32 * 16)
#define QSW_PER_BH  ((size_t)QTILES * 3 * 32 * 16)   // 603648
#define KSW_PER_BH  ((size_t)NTQ * 3 * 32 * 16)      // 39936
#define VSW_PER_BH  ((size_t)KSV * 6 * 32 * 16)      // 39936
#define ASW_ELEMS   ((size_t)MT2 * 6 * 32 * 16)

__device__ __forceinline__ v8f wmma_f16(v16h a, v16h b, v8f c) {
    return __builtin_amdgcn_wmma_f32_16x16x32_f16(false, a, false, b, (short)0, c, false, false);
}

// A-fragment layout inverse: given k' in [0,32) -> (hi lane bit, j element)
__device__ __forceinline__ void a_inv(int k, int& hi, int& j) {
    hi = (k >> 3) & 1;
    j  = (k & 16) ? 8 + (k & 7) : (k & 7);
}
// B-fragment layout inverse
__device__ __forceinline__ void b_inv(int k, int& hi, int& j) {
    hi = (k >> 3) & 1;
    j  = (k & 16) ? 8 + (k & 7) : (k & 7);
}

// ---------------- operand swizzle kernels ----------------

// x (MTOT1 x 96) f32 -> A-swizzled f16 [MT1][3][32][16]; pad rows zeroed
__global__ __launch_bounds__(256) void swizzleA_x(const float* __restrict__ src,
                                                  _Float16* __restrict__ dst) {
    int i = blockIdx.x * 256 + threadIdx.x;
    if (i >= MT1 * 16 * HDIM) return;
    int r = i / HDIM, c = i % HDIM;
    float v = (r < MTOT1) ? src[(size_t)r * HDIM + c] : 0.f;
    int mt = r >> 4, lo = r & 15;
    int ks = c >> 5, k = c & 31, hi, j;
    a_inv(k, hi, j);
    dst[(((size_t)mt * 3 + ks) * 32 + hi * 16 + lo) * 16 + j] = (_Float16)v;
}

// weight (nout x kin) f32 -> B-swizzled f16 [kin/32][nout/16][32][16] (B matrix = W^T, K x N)
__global__ __launch_bounds__(256) void swizzleB_w(const float* __restrict__ src,
                                                  _Float16* __restrict__ dst,
                                                  int nout, int kin) {
    int i = blockIdx.x * 256 + threadIdx.x;
    if (i >= nout * kin) return;
    int c = i / kin;      // N dim (output feature)
    int k = i % kin;      // K dim (input feature)
    int ks = k >> 5, kk = k & 31, hi, j;
    b_inv(kk, hi, j);
    int nt = c >> 4, lo = c & 15;
    int ntiles = nout >> 4;
    dst[(((size_t)ks * ntiles + nt) * 32 + hi * 16 + lo) * 16 + j] = (_Float16)src[i];
}

// ---------------- qkv GEMM: (100356 x 96) x (96 x 576), 1x4 tile blocking ----------------
// outputs q/k/v f32 laid out (MTOT1 rows, nh, 96): addr = r*192 + hh*96 + dd

__global__ __launch_bounds__(128) void qkv_gemm(const _Float16* __restrict__ xsw,
                                                const _Float16* __restrict__ wsw,
                                                const float* __restrict__ bias,
                                                float* __restrict__ qf,
                                                float* __restrict__ kf,
                                                float* __restrict__ vf) {
    int wave = blockIdx.x * 4 + (threadIdx.x >> 5);
    int lane = threadIdx.x & 31;
    if (wave >= MT1 * 9) return;
    int mt = wave / 9, ng = wave % 9;

    v8f a0 = {}, a1 = {}, a2 = {}, a3 = {};
#pragma unroll
    for (int ks = 0; ks < 3; ++ks) {
        v16h a = *(const v16h*)(xsw + (((size_t)mt * 3 + ks) * 32 + lane) * 16);
        const _Float16* wb = wsw + (((size_t)ks * 36 + ng * 4) * 32 + lane) * 16;
        a0 = wmma_f16(a, *(const v16h*)(wb), a0);
        a1 = wmma_f16(a, *(const v16h*)(wb + 512), a1);
        a2 = wmma_f16(a, *(const v16h*)(wb + 1024), a2);
        a3 = wmma_f16(a, *(const v16h*)(wb + 1536), a3);
    }

    v8f accs[4] = {a0, a1, a2, a3};
#pragma unroll
    for (int u = 0; u < 4; ++u) {
        int col   = (ng * 4 + u) * 16 + (lane & 15);
        int which = col / ODIM;
        int rem   = col % ODIM;
        float bv  = bias[col];
        float* dst = (which == 0) ? qf : ((which == 1) ? kf : vf);
        dst += rem; // hh*96 + dd folded
#pragma unroll
        for (int rv = 0; rv < 8; ++rv) {
            int r = mt * 16 + rv + ((lane & 16) ? 8 : 0);
            if (r < MTOT1) dst[(size_t)r * ODIM] = accs[u][rv] + bv;
        }
    }
}

// ---------------- depthwise 3x3x3 pool + LayerNorm ----------------
// src (MTOT1, 2, 96) f32. one block per output token, 96 threads = channels
// mode 0: q  -> A-swizzled q16sw (bh-major) + qf32p (bh, QN, 96) f32
// mode 1: kT -> B-swizzled k16sw  (K dim = channel, N dim = token, padded)
// mode 2: v  -> B-swizzled v16sw  (K dim = token, N dim = channel, padded)

__global__ __launch_bounds__(96) void pool_ln_kernel(const float* __restrict__ src,
                                                     const float* __restrict__ cw,  // (96,1,3,3,3)
                                                     const float* __restrict__ gamma,
                                                     const float* __restrict__ beta,
                                                     int mode, int oh, int ow, int sh, int sw,
                                                     int oNvalid, int padN,
                                                     _Float16* __restrict__ out16,
                                                     float* __restrict__ outf) {
    __shared__ float vals[96];
    int bh = blockIdx.x / padN;
    int n  = blockIdx.x % padN;
    int d  = threadIdx.x;
    int b_ = bh >> 1, hh = bh & 1;

    float y = 0.f;
    if (n < oNvalid) {
        const float* sb = src + ((size_t)b_ * N_IN) * ODIM + hh * HDIM + d;
        float v;
        if (n == 0) {
            v = sb[0]; // cls token
        } else {
            int s  = n - 1;
            int t2 = s / (oh * ow);
            int rr = s % (oh * ow);
            int y2 = rr / ow;
            int x2 = rr % ow;
            float acc = 0.f;
#pragma unroll
            for (int dt = 0; dt < 3; ++dt)
#pragma unroll
                for (int dh = 0; dh < 3; ++dh)
#pragma unroll
                    for (int dw = 0; dw < 3; ++dw) {
                        int ti = t2 + dt - 1;
                        int yi = y2 * sh + dh - 1;
                        int xi = x2 * sw + dw - 1;
                        if (ti >= 0 && ti < T_IN && yi >= 0 && yi < H_IN && xi >= 0 && xi < W_IN) {
                            size_t tok = 1 + ((size_t)ti * H_IN + yi) * W_IN + xi;
                            acc += sb[tok * ODIM] * cw[d * 27 + dt * 9 + dh * 3 + dw];
                        }
                    }
            v = acc;
        }
        vals[d] = v;
        __syncthreads();
        float m = 0.f;
#pragma unroll 8
        for (int i = 0; i < 96; ++i) m += vals[i];
        m *= (1.f / 96.f);
        float var = 0.f;
#pragma unroll 8
        for (int i = 0; i < 96; ++i) { float t = vals[i] - m; var += t * t; }
        var *= (1.f / 96.f);
        y = (v - m) * rsqrtf(var + 1e-5f) * gamma[d] + beta[d];
    }

    if (mode == 0) {
        int mt = n >> 4, lo = n & 15;
        int ks = d >> 5, k = d & 31, hi, j;
        a_inv(k, hi, j);
        out16[(size_t)bh * QSW_PER_BH + (((size_t)mt * 3 + ks) * 32 + hi * 16 + lo) * 16 + j] = (_Float16)y;
        outf[((size_t)bh * QN + n) * HDIM + d] = y;
    } else if (mode == 1) {
        // B for S=Q K^T: K dim = channel d (3 ks), N dim = token n (26 tiles)
        int ks = d >> 5, kk = d & 31, hi, j;
        b_inv(kk, hi, j);
        int nt = n >> 4, lo = n & 15;
        out16[(size_t)bh * KSW_PER_BH + (((size_t)nt * 3 + ks) * 32 + hi * 16 + lo) * 16 + j] = (_Float16)y;
    } else {
        // B for O=P V: K dim = token n (13 ks), N dim = channel d (6 tiles)
        int ks = n >> 5, kk = n & 31, hi, j;
        b_inv(kk, hi, j);
        int nt = d >> 4, lo = d & 15;
        out16[(size_t)bh * VSW_PER_BH + (((size_t)ks * 6 + nt) * 32 + hi * 16 + lo) * 16 + j] = (_Float16)y;
    }
}

// ---------------- fused attention: one wave per (bh, 16-row q tile) ----------------

__global__ __launch_bounds__(32) void attn_fused(const _Float16* __restrict__ q16sw,
                                                 const float* __restrict__ qf32,   // (bh, QN, 96)
                                                 const _Float16* __restrict__ k16sw,
                                                 const _Float16* __restrict__ v16sw,
                                                 const float* __restrict__ rph,    // (55, 96)
                                                 const float* __restrict__ rpw,    // (55, 96)
                                                 const float* __restrict__ rpt,    // (15, 96)
                                                 _Float16* __restrict__ attsw) {   // A-swizzled (MT2,6,32,16)
    __shared__ float S[16][KNP];
    __shared__ __align__(32) _Float16 Psw[KSV][32][16];
    __shared__ float relT[16][8];
    __shared__ float relH[16][8];
    __shared__ float relW[16][8];

    int bh   = blockIdx.x / QTILES;
    int qt   = blockIdx.x % QTILES;
    int lane = threadIdx.x;
    int q0   = qt * 16;

    const _Float16* qbase = q16sw + (size_t)bh * QSW_PER_BH + ((size_t)qt * 3) * 512;
    const _Float16* kbase = k16sw + (size_t)bh * KSW_PER_BH;
    const _Float16* vbase = v16sw + (size_t)bh * VSW_PER_BH;
    const float*    qfb   = qf32 + (size_t)bh * QN * HDIM;

    // Q fragments (3 K-steps)
    v16h aq0 = *(const v16h*)(qbase + (size_t)lane * 16);
    v16h aq1 = *(const v16h*)(qbase + 512 + (size_t)lane * 16);
    v16h aq2 = *(const v16h*)(qbase + 1024 + (size_t)lane * 16);

    // --- S = scale * Q K^T ---
    for (int nt = 0; nt < NTQ; ++nt) {
        const _Float16* kb = kbase + ((size_t)nt * 3) * 512 + (size_t)lane * 16;
        v8f c = {};
        c = wmma_f16(aq0, *(const v16h*)(kb), c);
        c = wmma_f16(aq1, *(const v16h*)(kb + 512), c);
        c = wmma_f16(aq2, *(const v16h*)(kb + 1024), c);
        int colj = nt * 16 + (lane & 15);
#pragma unroll
        for (int rv = 0; rv < 8; ++rv)
            S[rv + ((lane & 16) ? 8 : 0)][colj] = c[rv] * SCALE_F;
    }
    __syncthreads();

    // --- decomposed rel-pos: 22 dots (8 t + 7 h + 7 w) per q row ---
    for (int task = lane; task < 16 * 22; task += 32) {
        int m = task / 22, wsel = task % 22;
        int r = q0 + m;
        float dv = 0.f;
        if (r >= 1 && r < QN) {
            int s  = r - 1;
            int t2 = s / (QH * QW);
            int rr = s % (QH * QW);
            int y2 = rr / QW;
            int x2 = rr % QW;
            const float* tb;
            if (wsel < 8)       tb = rpt + (size_t)(t2 - wsel + 7) * HDIM;             // t - kt + 7
            else if (wsel < 15) tb = rph + (size_t)(y2 - 4 * (wsel - 8) + 24) * HDIM;  // y - 4*kh + 24
            else                tb = rpw + (size_t)(x2 - 4 * (wsel - 15) + 24) * HDIM; // x - 4*kw + 24
            const float* qr = qfb + (size_t)r * HDIM;
#pragma unroll 8
            for (int c = 0; c < HDIM; ++c) dv += qr[c] * tb[c];
        }
        if (wsel < 8)       relT[m][wsel] = dv;
        else if (wsel < 15) relH[m][wsel - 8] = dv;
        else                relW[m][wsel - 15] = dv;
    }
    __syncthreads();

    // --- add rel to S[:, 1:] ---
    for (int e = lane; e < 16 * (KN - 1); e += 32) {
        int m = e / (KN - 1);
        int j = e % (KN - 1);
        int kt = j / (KH * KW);
        int rr = j % (KH * KW);
        S[m][1 + j] += relT[m][kt] + relH[m][rr / KW] + relW[m][rr % KW];
    }
    __syncthreads();

    // --- softmax per row; write probs directly in A-fragment LDS layout ---
    if (lane < 16) {
        int r = q0 + lane;
        float inv = 0.f;
        if (r < QN) {
            float mx = -3.4e38f;
            for (int j = 0; j < KN; ++j) mx = fmaxf(mx, S[lane][j]);
            float sum = 0.f;
            for (int j = 0; j < KN; ++j) {
                float e = __expf(S[lane][j] - mx);
                S[lane][j] = e;
                sum += e;
            }
            inv = 1.f / sum;
        }
        for (int jc = 0; jc < KNP; ++jc) {
            float val = (jc < KN) ? S[lane][jc] * inv : 0.f;
            int ks = jc >> 5, kk = jc & 31, hi, jj;
            a_inv(kk, hi, jj);
            Psw[ks][hi * 16 + lane][jj] = (_Float16)val;
        }
    }
    __syncthreads();

    // --- O = P V + residual(q), store A-swizzled for proj GEMM ---
    int b_ = bh >> 1, hh = bh & 1;
    for (int ntile = 0; ntile < HDIM / 16; ++ntile) {
        v8f c = {};
#pragma unroll
        for (int ks = 0; ks < KSV; ++ks) {
            v16h a = *(const v16h*)&Psw[ks][lane][0];
            v16h b = *(const v16h*)(vbase + (((size_t)ks * 6 + ntile) * 32 + lane) * 16);
            c = wmma_f16(a, b, c);
        }
        int dd  = ntile * 16 + (lane & 15);
        int col = hh * HDIM + dd;
        int ks2 = col >> 5, kk = col & 31, hi2, jj;
        a_inv(kk, hi2, jj);
#pragma unroll
        for (int rv = 0; rv < 8; ++rv) {
            int m = rv + ((lane & 16) ? 8 : 0);
            int r = q0 + m;
            if (r < QN) {
                float o  = c[rv] + qfb[(size_t)r * HDIM + dd];
                int rg   = b_ * QN + r;
                int mt   = rg >> 4, lo = rg & 15;
                attsw[(((size_t)mt * 6 + ks2) * 32 + hi2 * 16 + lo) * 16 + jj] = (_Float16)o;
            }
        }
    }
}

// ---------------- output projection: (25092 x 192) x (192 x 192) + bias, 1x4 blocking --------

__global__ __launch_bounds__(128) void proj_gemm(const _Float16* __restrict__ asw,
                                                 const _Float16* __restrict__ wsw,
                                                 const float* __restrict__ bias,
                                                 float* __restrict__ out) {
    int wave = blockIdx.x * 4 + (threadIdx.x >> 5);
    int lane = threadIdx.x & 31;
    if (wave >= MT2 * 3) return;
    int mt = wave / 3, ng = wave % 3;

    v8f a0 = {}, a1 = {}, a2 = {}, a3 = {};
#pragma unroll
    for (int ks = 0; ks < 6; ++ks) {
        v16h a = *(const v16h*)(asw + (((size_t)mt * 6 + ks) * 32 + lane) * 16);
        const _Float16* wb = wsw + (((size_t)ks * 12 + ng * 4) * 32 + lane) * 16;
        a0 = wmma_f16(a, *(const v16h*)(wb), a0);
        a1 = wmma_f16(a, *(const v16h*)(wb + 512), a1);
        a2 = wmma_f16(a, *(const v16h*)(wb + 1024), a2);
        a3 = wmma_f16(a, *(const v16h*)(wb + 1536), a3);
    }

    v8f accs[4] = {a0, a1, a2, a3};
#pragma unroll
    for (int u = 0; u < 4; ++u) {
        int col  = (ng * 4 + u) * 16 + (lane & 15);
        float bv = bias[col];
#pragma unroll
        for (int rv = 0; rv < 8; ++rv) {
            int r = mt * 16 + rv + ((lane & 16) ? 8 : 0);
            if (r < MTOT2) out[(size_t)r * ODIM + col] = accs[u][rv] + bv;
        }
    }
}

// ---------------- host launcher ----------------

extern "C" void kernel_launch(void* const* d_in, const int* in_sizes, int n_in,
                              void* d_out, int out_size, void* d_ws, size_t ws_size,
                              hipStream_t stream) {
    const float* x      = (const float*)d_in[0];
    const float* qkv_w  = (const float*)d_in[1];
    const float* qkv_b  = (const float*)d_in[2];
    const float* proj_w = (const float*)d_in[3];
    const float* proj_b = (const float*)d_in[4];
    const float* pq_w   = (const float*)d_in[5];
    const float* nq_g   = (const float*)d_in[6];
    const float* nq_b   = (const float*)d_in[7];
    const float* pk_w   = (const float*)d_in[8];
    const float* nk_g   = (const float*)d_in[9];
    const float* nk_b   = (const float*)d_in[10];
    const float* pv_w   = (const float*)d_in[11];
    const float* nv_g   = (const float*)d_in[12];
    const float* nv_b   = (const float*)d_in[13];
    const float* rph    = (const float*)d_in[14];
    const float* rpw    = (const float*)d_in[15];
    const float* rpt    = (const float*)d_in[16];
    float* out = (float*)d_out;

    char* ws = (char*)d_ws;
    size_t off = 0;
    auto alloc = [&](size_t bytes) -> void* {
        void* p = ws + off;
        off += (bytes + 255) & ~(size_t)255;
        return p;
    };

    _Float16* xsw    = (_Float16*)alloc(XSW_ELEMS * 2);
    _Float16* qkvwsw = (_Float16*)alloc((size_t)3 * 36 * 32 * 16 * 2);
    _Float16* prjwsw = (_Float16*)alloc((size_t)6 * 12 * 32 * 16 * 2);
    float*    qf     = (float*)alloc((size_t)MTOT1 * ODIM * 4);
    float*    kf     = (float*)alloc((size_t)MTOT1 * ODIM * 4);
    float*    vf     = (float*)alloc((size_t)MTOT1 * ODIM * 4);
    _Float16* q16sw  = (_Float16*)alloc((size_t)8 * QSW_PER_BH * 2);
    float*    qf32p  = (float*)alloc((size_t)8 * QN * HDIM * 4);
    _Float16* k16sw  = (_Float16*)alloc((size_t)8 * KSW_PER_BH * 2);
    _Float16* v16sw  = (_Float16*)alloc((size_t)8 * VSW_PER_BH * 2);
    _Float16* attsw  = (_Float16*)alloc(ASW_ELEMS * 2);
    (void)ws_size; (void)in_sizes; (void)n_in; (void)out_size;

    // 1. operand swizzles
    {
        int n = MT1 * 16 * HDIM;
        swizzleA_x<<<(n + 255) / 256, 256, 0, stream>>>(x, xsw);
        swizzleB_w<<<(576 * 96 + 255) / 256, 256, 0, stream>>>(qkv_w, qkvwsw, 576, 96);
        swizzleB_w<<<(192 * 192 + 255) / 256, 256, 0, stream>>>(proj_w, prjwsw, 192, 192);
    }
    // 2. qkv GEMM
    {
        int waves = MT1 * 9;
        qkv_gemm<<<(waves + 3) / 4, 128, 0, stream>>>(xsw, qkvwsw, qkv_b, qf, kf, vf);
    }
    // 3. pools + layernorm
    pool_ln_kernel<<<8 * QN, 96, 0, stream>>>(qf, pq_w, nq_g, nq_b, 0, QH, QW, 2, 2, QN, QN, q16sw, qf32p);
    pool_ln_kernel<<<8 * KNP, 96, 0, stream>>>(kf, pk_w, nk_g, nk_b, 1, KH, KW, 8, 8, KN, KNP, k16sw, nullptr);
    pool_ln_kernel<<<8 * KNP, 96, 0, stream>>>(vf, pv_w, nv_g, nv_b, 2, KH, KW, 8, 8, KN, KNP, v16sw, nullptr);
    // 4. fused attention
    attn_fused<<<8 * QTILES, 32, 0, stream>>>(q16sw, qf32p, k16sw, v16sw, rph, rpw, rpt, attsw);
    // 5. projection
    {
        int waves = MT2 * 3;
        proj_gemm<<<(waves + 3) / 4, 128, 0, stream>>>(attsw, prjwsw, proj_b, out);
    }
}
